// RelationalModule_52132313039001
// MI455X (gfx1250) — matrix operand order repaired and checked by reading, not compiled, not run on hardware
//
#include <hip/hip_runtime.h>
#include <hip/hip_bf16.h>

typedef __attribute__((ext_vector_type(16))) _Float16 v16h;
typedef __attribute__((ext_vector_type(8)))  _Float16 v8h;
typedef __attribute__((ext_vector_type(4)))  _Float16 v4h;
typedef __attribute__((ext_vector_type(8)))  float    v8f;

#define RN_B 32
#define RN_N 64
#define RN_D 128
#define RN_H 256
#define NPAD 264   // 256 halfs + 8 pad -> row stride 528 B (16B aligned, bank-staggered)

// ---------------------------------------------------------------------------
// Kernel 1: pack gW2 (f32, [K=256][N=256] row-major) into WMMA B-fragment
// layout for v_wmma_f32_16x16x32_f16:
//   fragment f = nt*8 + kf  (nt = N-tile 0..15, kf = K-frag 0..7)
//   lane l: n = nt*16 + (l%16); element e (0..15): K = kf*32 + (l/16)*16 + e
// Stored as halfs at index f*512 + l*16 + e  (each lane's 16 halfs contiguous)
// ---------------------------------------------------------------------------
__global__ __launch_bounds__(256) void rn_pack_w2(const float* __restrict__ gW2,
                                                  _Float16* __restrict__ W2p) {
    const int idx = blockIdx.x * 256 + threadIdx.x;   // 0..65535
    const int e  = idx & 15;
    const int l  = (idx >> 4) & 31;
    const int f  = idx >> 9;                          // 0..127
    const int nt = f >> 3;
    const int kf = f & 7;
    const int n  = nt * 16 + (l & 15);
    const int K  = kf * 32 + (l >> 4) * 16 + e;
    W2p[idx] = (_Float16)gW2[K * RN_H + n];
}

// ---------------------------------------------------------------------------
// Kernel 2: L = obj @ gW1[0:128,:] + gb1 ;  R = obj @ gW1[128:256,:]
// One block per (b,n) row of objects (2048 blocks), 256 threads (one per h).
// ---------------------------------------------------------------------------
__global__ __launch_bounds__(256) void rn_lr(const float* __restrict__ objects,
                                             const float* __restrict__ gW1,
                                             const float* __restrict__ gb1,
                                             float* __restrict__ L,
                                             float* __restrict__ R) {
    __shared__ float so[RN_D];
    const int row = blockIdx.x;          // 0..2047 = b*64 + n
    const int t   = threadIdx.x;         // output channel
    if (t < RN_D) so[t] = objects[(size_t)row * RN_D + t];
    __syncthreads();
    float accL = gb1[t];
    float accR = 0.0f;
#pragma unroll 8
    for (int k = 0; k < RN_D; ++k) {
        const float o = so[k];
        accL += o * gW1[k * RN_H + t];
        accR += o * gW1[(RN_D + k) * RN_H + t];
    }
    L[(size_t)row * RN_H + t] = accL;
    R[(size_t)row * RN_H + t] = accR;
}

// ---------------------------------------------------------------------------
// Kernel 3 (the hot one): per block = one (b, i).
//   Phase 1: sA[j][k] = (f16) relu(L[b,i,k] + R[b,j,k])   (64 x 256, LDS)
//   Phase 2: S = relu(sA @ gW2 + gb2), masked (j != i), column-summed,
//            atomically accumulated into agg[b][256].
// 8 waves; wave w owns N-tiles {2w, 2w+1}; B-frags held in registers.
// ---------------------------------------------------------------------------
__global__ __launch_bounds__(256) void rn_g_main(const float* __restrict__ L,
                                                 const float* __restrict__ R,
                                                 const v16h* __restrict__ W2p,
                                                 const float* __restrict__ gb2,
                                                 float* __restrict__ agg) {
    __shared__ __align__(16) _Float16 sA[RN_N * NPAD];
    const int b = blockIdx.x >> 6;   // 0..31
    const int i = blockIdx.x & 63;   // 0..63
    const int t = threadIdx.x;

    // ---- Phase 1: build A = relu(L[i] + R[j]) in f16 ----
    {
        const int j  = t >> 2;              // 0..63
        const int kc = (t & 3) << 6;        // 0,64,128,192
        const float* Lp = L + (((size_t)b * RN_N + i) << 8) + kc;
        const float* Rp = R + (((size_t)b * RN_N + j) << 8) + kc;
        _Float16* dst = &sA[j * NPAD + kc];
#pragma unroll
        for (int k = 0; k < 64; k += 4) {
            const float4 lv = *(const float4*)(Lp + k);
            const float4 rv = *(const float4*)(Rp + k);
            v4h hv;
            hv[0] = (_Float16)fmaxf(lv.x + rv.x, 0.0f);
            hv[1] = (_Float16)fmaxf(lv.y + rv.y, 0.0f);
            hv[2] = (_Float16)fmaxf(lv.z + rv.z, 0.0f);
            hv[3] = (_Float16)fmaxf(lv.w + rv.w, 0.0f);
            *(v4h*)(dst + k) = hv;
        }
    }
    __syncthreads();

    // ---- Phase 2: WMMA GEMM + bias + relu + masked reduction ----
    const int w  = t >> 5;    // wave 0..7
    const int l  = t & 31;    // lane
    const int lm = l & 15;    // A row within tile / C column
    const int lh = l >> 4;    // lane half

#pragma unroll
    for (int ntl = 0; ntl < 2; ++ntl) {
        const int nt = w * 2 + ntl;          // N-tile 0..15
        const int n  = nt * 16 + lm;         // output channel for this lane

        // hoist the 8 K-fragments of gW2 for this N-tile into registers
        v16h bf[8];
#pragma unroll
        for (int kf = 0; kf < 8; ++kf)
            bf[kf] = W2p[(nt * 8 + kf) * 32 + l];

        const float bias = gb2[n];
        float colsum = 0.0f;

#pragma unroll
        for (int mt = 0; mt < 4; ++mt) {
            v8f c = {};
            const _Float16* arow = &sA[(mt * 16 + lm) * NPAD];
#pragma unroll
            for (int kf = 0; kf < 8; ++kf) {
                union { v16h v; v8h h[2]; } a;
                // A layout: K = kf*32 + lh*16 + e
                a.h[0] = *(const v8h*)(arow + kf * 32 + lh * 16);
                a.h[1] = *(const v8h*)(arow + kf * 32 + lh * 16 + 8);
                c = __builtin_amdgcn_wmma_f32_16x16x32_f16(
                        false, a.v, false, bf[kf], (short)0, c, false, false);
            }
            // C layout: c[r] = row (lh ? 8+r : r) of this M-tile, column lm
#pragma unroll
            for (int r = 0; r < 8; ++r) {
                const int jrow = mt * 16 + (lh ? 8 + r : r);
                float v = fmaxf(c[r] + bias, 0.0f);
                colsum += (jrow != i) ? v : 0.0f;
            }
        }
        // lanes l and l^16 hold partial sums for the same column -> combine
        colsum += __shfl_xor(colsum, 16, 32);
        if (lh == 0) atomicAdd(&agg[b * RN_H + n], colsum);
    }
}

// ---------------------------------------------------------------------------
// Kernel 4: tail.  aggH = agg @ gW3 + 4032*gb3 ; out = relu(aggH@fW1+fb1)@fW2+fb2
// One block per batch row, 256 threads (one per channel).
// ---------------------------------------------------------------------------
__global__ __launch_bounds__(256) void rn_tail(const float* __restrict__ agg,
                                               const float* __restrict__ gW3,
                                               const float* __restrict__ gb3,
                                               const float* __restrict__ fW1,
                                               const float* __restrict__ fb1,
                                               const float* __restrict__ fW2,
                                               const float* __restrict__ fb2,
                                               float* __restrict__ out) {
    __shared__ float s0[RN_H];
    __shared__ float s1[RN_H];
    const int b = blockIdx.x;
    const int t = threadIdx.x;
    s0[t] = agg[b * RN_H + t];
    __syncthreads();
    // layer g3 (linear map commutes with the pair-sum; 4032 = 64*63 pairs)
    float acc = 4032.0f * gb3[t];
#pragma unroll 8
    for (int k = 0; k < RN_H; ++k) acc += s0[k] * gW3[k * RN_H + t];
    s1[t] = acc;
    __syncthreads();
    // f layer 1 + relu
    acc = fb1[t];
#pragma unroll 8
    for (int k = 0; k < RN_H; ++k) acc += s1[k] * fW1[k * RN_H + t];
    s0[t] = fmaxf(acc, 0.0f);
    __syncthreads();
    // f layer 2
    acc = fb2[t];
#pragma unroll 8
    for (int k = 0; k < RN_H; ++k) acc += s0[k] * fW2[k * RN_H + t];
    out[b * RN_H + t] = acc;
}

// ---------------------------------------------------------------------------
extern "C" void kernel_launch(void* const* d_in, const int* in_sizes, int n_in,
                              void* d_out, int out_size, void* d_ws, size_t ws_size,
                              hipStream_t stream) {
    const float* objects = (const float*)d_in[0];
    const float* gW1 = (const float*)d_in[1];
    const float* gb1 = (const float*)d_in[2];
    const float* gW2 = (const float*)d_in[3];
    const float* gb2 = (const float*)d_in[4];
    const float* gW3 = (const float*)d_in[5];
    const float* gb3 = (const float*)d_in[6];
    const float* fW1 = (const float*)d_in[7];
    const float* fb1 = (const float*)d_in[8];
    const float* fW2 = (const float*)d_in[9];
    const float* fb2 = (const float*)d_in[10];
    float* out = (float*)d_out;

    // workspace layout
    char* ws = (char*)d_ws;
    float*    agg = (float*)ws;                               //  32 KB
    float*    Lb  = (float*)(ws + (32 * 1024));               //   2 MB
    float*    Rb  = (float*)(ws + (32 * 1024) + (2u << 20));  //   2 MB
    _Float16* W2p = (_Float16*)(ws + (32 * 1024) + (4u << 20)); // 128 KB

    // agg must be zero every launch (harness does not re-poison/clear)
    hipMemsetAsync(agg, 0, RN_B * RN_H * sizeof(float), stream);

    rn_pack_w2<<<256, 256, 0, stream>>>(gW2, W2p);
    rn_lr<<<RN_B * RN_N, 256, 0, stream>>>(objects, gW1, gb1, Lb, Rb);
    rn_g_main<<<RN_B * RN_N, 256, 0, stream>>>(Lb, Rb, (const v16h*)W2p, gb2, agg);
    rn_tail<<<RN_B, 256, 0, stream>>>(agg, gW3, gb3, fW1, fb1, fW2, fb2, out);
}
